// MultiHeadAttention_3255585210398
// MI455X (gfx1250) — matrix operand order repaired and checked by reading, not compiled, not run on hardware
//
#include <hip/hip_runtime.h>
#include <hip/hip_bf16.h>

// ---------------------------------------------------------------------------
// MultiHeadAttention for gfx1250 (MI455X): bf16 WMMA flash attention with
// async global->LDS double buffering (ASYNCcnt) for K/V strips.
//   B=4, S=2048, D=1024, H=16, DQ=DV=64
// ---------------------------------------------------------------------------

#define B_   4
#define S_   2048
#define D_   1024
#define H_   16
#define E_   64          // head dim (DQ == DV)

// padded LDS row strides (elements) - chosen so row_stride mod 64 banks has no
// factor hitting lane deltas < 16 => conflict-free fragment gathers
#define KLD_ 72          // 32 K-rows * 144B
#define VLD_ 40          // 64 V-rows *  80B
#define PLD_ 40          // 16 P-rows *  80B

typedef __attribute__((ext_vector_type(16))) __bf16 v16bf;
typedef __attribute__((ext_vector_type(8)))  float  v8f;

union ABFrag { v16bf v; unsigned int u[8]; };
union CDFrag { v8f   v; float        f[8]; };

// float -> bf16 (round-to-nearest-even), as raw bits
__device__ __forceinline__ unsigned short f2bf(float f) {
    unsigned int u = __float_as_uint(f);
    u += 0x7FFFu + ((u >> 16) & 1u);
    return (unsigned short)(u >> 16);
}
__device__ __forceinline__ float bf2f(unsigned int lo16) {
    return __uint_as_float(lo16 << 16);
}

// ---------------------------------------------------------------------------
// Fragment loaders (see round-0 derivation from ISA VGPR tables).
// A-frag: row = lane%16, element e -> k = e + (e>=8?8:0) + (lane>=16?8:0)
// B-frag from transposed tile Bt[n][k]: col n = lane%16, element e -> k = e + (e>=8?8:0) + (lane>=16?8:0)
// C/D-frag: element i -> row i + (lane>=16?8:0), col = lane%16
// ---------------------------------------------------------------------------
__device__ __forceinline__ void load_a_frag(ABFrag& a, const unsigned short* src,
                                            int ld, int lane) {
    const int row = lane & 15;
    const int hi8 = (lane >> 4) ? 8 : 0;
    const unsigned short* p = src + (size_t)row * ld;
#pragma unroll
    for (int v = 0; v < 8; ++v) {
        int e  = 2 * v;
        int k0 = e + ((e >= 8) ? 8 : 0) + hi8;
        a.u[v] = *(const unsigned int*)(p + k0);
    }
}

__device__ __forceinline__ void load_bt_frag(ABFrag& b, const unsigned short* src,
                                             int ld, int lane) {
    const int n    = lane & 15;
    const int hi16 = (lane >> 4) ? 8 : 0;
    const unsigned short* p = src + (size_t)n * ld;
#pragma unroll
    for (int v = 0; v < 8; ++v) {
        int k0 = 2 * v + ((2 * v >= 8) ? 8 : 0) + hi16;
        b.u[v] = *(const unsigned int*)(p + k0);
    }
}

__device__ __forceinline__ v8f wmma_bf16(const ABFrag& a, const ABFrag& b, v8f c) {
    return __builtin_amdgcn_wmma_f32_16x16x32_bf16(
        false, a.v, false, b.v, (short)0, c, false, false);
}

// async 16B global->LDS copy (per-lane addresses), tracked by ASYNCcnt
__device__ __forceinline__ void async_b128(unsigned lds_addr, unsigned long long gaddr) {
    asm volatile("global_load_async_to_lds_b128 %0, %1, off"
                 :: "v"(lds_addr), "v"(gaddr) : "memory");
}

// ---------------------------------------------------------------------------
// Conversion kernels
// ---------------------------------------------------------------------------
__global__ void cvt_f32_to_bf16(const float* __restrict__ in,
                                unsigned short* __restrict__ out, int n) {
    int i = blockIdx.x * blockDim.x + threadIdx.x;
    if (i < n) out[i] = f2bf(in[i]);
}

// in: W[h][d][e]  ->  out: Wt[h][e][d]   (bf16)
__global__ void cvt_w_transpose(const float* __restrict__ in,
                                unsigned short* __restrict__ out) {
    int i = blockIdx.x * blockDim.x + threadIdx.x;
    const int n = H_ * D_ * E_;
    if (i >= n) return;
    int e = i % E_;
    int d = (i / E_) % D_;
    int h = i / (E_ * D_);
    out[((size_t)h * E_ + e) * D_ + d] = f2bf(in[i]);
}

// in: Wo[k][n] (k=H*DV=1024, n=D=1024) -> out: WoT[n][k]  (bf16)
__global__ void cvt_wo_transpose(const float* __restrict__ in,
                                 unsigned short* __restrict__ out) {
    int i = blockIdx.x * blockDim.x + threadIdx.x;
    const int K = H_ * E_, N = D_;
    if (i >= K * N) return;
    int n = i % N;
    int k = i / N;
    out[(size_t)n * K + k] = f2bf(in[i]);
}

// ---------------------------------------------------------------------------
// QKV projection: one wave per (b, h, mat, 16-row s-block) computing a 16x64
// tile. Register-double-buffered K-loop (step 64 = 2x32) so loads for step
// i+1 overlap the WMMAs of step i.
// ---------------------------------------------------------------------------
__global__ __launch_bounds__(32)
void qkv_proj(const unsigned short* __restrict__ xb,
              const unsigned short* __restrict__ Wt,   // [3][H][E][D]
              unsigned short* __restrict__ Qb,
              unsigned short* __restrict__ Kb,
              unsigned short* __restrict__ Vt) {
    const int lane = threadIdx.x & 31;
    const int sblk = blockIdx.x;
    const int h    = blockIdx.y;
    const int bz   = blockIdx.z;
    const int b    = bz / 3;
    const int mat  = bz % 3;

    const unsigned short* xbase = xb + ((size_t)b * S_ + (size_t)sblk * 16) * D_;
    const unsigned short* wbase = Wt + ((size_t)mat * H_ + h) * (size_t)E_ * D_;

    CDFrag acc[4];
#pragma unroll
    for (int f = 0; f < 4; ++f)
#pragma unroll
        for (int i = 0; i < 8; ++i) acc[f].f[i] = 0.0f;

    ABFrag a0, a1, b0[4], b1[4];
    load_a_frag(a0, xbase, D_, lane);
#pragma unroll
    for (int nb = 0; nb < 4; ++nb)
        load_bt_frag(b0[nb], wbase + (size_t)(nb * 16) * D_, D_, lane);

    for (int d = 0; d < D_; d += 64) {
        // prefetch d+32 into buffer 1
        load_a_frag(a1, xbase + d + 32, D_, lane);
#pragma unroll
        for (int nb = 0; nb < 4; ++nb)
            load_bt_frag(b1[nb], wbase + (size_t)(nb * 16) * D_ + d + 32, D_, lane);
        // consume buffer 0 (depth d)
#pragma unroll
        for (int nb = 0; nb < 4; ++nb) acc[nb].v = wmma_bf16(a0, b0[nb], acc[nb].v);
        // prefetch d+64 into buffer 0
        if (d + 64 < D_) {
            load_a_frag(a0, xbase + d + 64, D_, lane);
#pragma unroll
            for (int nb = 0; nb < 4; ++nb)
                load_bt_frag(b0[nb], wbase + (size_t)(nb * 16) * D_ + d + 64, D_, lane);
        }
        // consume buffer 1 (depth d+32)
#pragma unroll
        for (int nb = 0; nb < 4; ++nb) acc[nb].v = wmma_bf16(a1, b1[nb], acc[nb].v);
    }

    const int col   = lane & 15;
    const int rbase = (lane >> 4) ? 8 : 0;
    const size_t bh = (size_t)b * H_ + h;
#pragma unroll
    for (int nb = 0; nb < 4; ++nb)
#pragma unroll
        for (int i = 0; i < 8; ++i) {
            int s = sblk * 16 + rbase + i;
            int e = nb * 16 + col;
            unsigned short val = f2bf(acc[nb].f[i]);
            if (mat == 0)      Qb[(bh * S_ + s) * E_ + e] = val;
            else if (mat == 1) Kb[(bh * S_ + s) * E_ + e] = val;
            else               Vt[(bh * E_ + e) * S_ + s] = val;
        }
}

// ---------------------------------------------------------------------------
// Flash attention: one wave per (b, h, 16-query block); online softmax over
// S=2048 keys in 32-key steps. K/V strips are staged in LDS with async
// global->LDS b128 copies, double-buffered on ASYNCcnt.
// ---------------------------------------------------------------------------
__global__ __launch_bounds__(32)
void flash_attn(const unsigned short* __restrict__ Qb,
                const unsigned short* __restrict__ Kb,
                const unsigned short* __restrict__ Vt,
                unsigned short* __restrict__ att) {
    __shared__ unsigned short Klds[2][32 * KLD_];   // 32 rows x 128B (+16B pad)
    __shared__ unsigned short Vlds[2][64 * VLD_];   // 64 rows x  64B (+16B pad)
    __shared__ unsigned short Plds[16 * PLD_];      // 16 rows x  64B (+16B pad)

    const int lane = threadIdx.x & 31;
    const int sblk = blockIdx.x;
    const int h    = blockIdx.y;
    const int b    = blockIdx.z;

    const size_t bh = (size_t)b * H_ + h;
    const unsigned short* Qbase = Qb + (bh * S_ + (size_t)sblk * 16) * E_;
    const unsigned short* Kbase = Kb + bh * (size_t)S_ * E_;
    const unsigned short* Vbase = Vt + bh * (size_t)E_ * S_;

    const unsigned kofs[2] = { (unsigned)(size_t)&Klds[0][0], (unsigned)(size_t)&Klds[1][0] };
    const unsigned vofs[2] = { (unsigned)(size_t)&Vlds[0][0], (unsigned)(size_t)&Vlds[1][0] };
    const unsigned long long KgB = (unsigned long long)(size_t)Kbase;
    const unsigned long long VgB = (unsigned long long)(size_t)Vbase;

    // issue one 32-key strip (K: 4KB contiguous, V: 64 rows x 64B) -> 16 asyncs
    auto issue_strip = [&](int t, int buf) {
        unsigned long long kg = KgB + (unsigned long long)t * (E_ * 2);
#pragma unroll
        for (int j = 0; j < 8; ++j) {
            int tt = j * 4 + (lane >> 3);
            int c  = lane & 7;
            async_b128(kofs[buf] + (unsigned)(tt * (KLD_ * 2) + c * 16),
                       kg + (unsigned long long)(tt * 128 + c * 16));
        }
#pragma unroll
        for (int j = 0; j < 8; ++j) {
            int vr = j * 8 + (lane >> 2);
            int c  = lane & 3;
            async_b128(vofs[buf] + (unsigned)(vr * (VLD_ * 2) + c * 16),
                       VgB + ((unsigned long long)vr * S_ + (unsigned long long)t) * 2 + c * 16);
        }
    };

    // Load Q fragments (2 over head-dim), pre-scaled by 1/sqrt(64) = 0.125
    ABFrag qf[2];
#pragma unroll
    for (int df = 0; df < 2; ++df) {
        load_a_frag(qf[df], Qbase + df * 32, E_, lane);
#pragma unroll
        for (int v = 0; v < 8; ++v) {
            unsigned int u = qf[df].u[v];
            float lo = bf2f(u & 0xFFFFu) * 0.125f;
            float hi = bf2f(u >> 16)     * 0.125f;
            qf[df].u[v] = (unsigned int)f2bf(lo) | ((unsigned int)f2bf(hi) << 16);
        }
    }

    CDFrag acc[4];
#pragma unroll
    for (int f = 0; f < 4; ++f)
#pragma unroll
        for (int i = 0; i < 8; ++i) acc[f].f[i] = 0.0f;

    float m[8], l[8];
#pragma unroll
    for (int i = 0; i < 8; ++i) { m[i] = -1e30f; l[i] = 0.0f; }

    const int col   = lane & 15;
    const int rbase = (lane >> 4) ? 8 : 0;

    issue_strip(0, 0);

    for (int t = 0; t < S_; t += 32) {
        const int cur = (t >> 5) & 1;
        if (t + 32 < S_) {
            issue_strip(t + 32, cur ^ 1);
            // <=16 outstanding => previous (current buffer's) batch has landed
            asm volatile("s_wait_asynccnt 0x10" ::: "memory");
        } else {
            asm volatile("s_wait_asynccnt 0x0" ::: "memory");
        }
        const unsigned short* Kst = &Klds[cur][0];
        const unsigned short* Vst = &Vlds[cur][0];

        // ---- scores: two 16x16 fragments over this 32-key strip ----
        CDFrag s0, s1;
#pragma unroll
        for (int i = 0; i < 8; ++i) { s0.f[i] = 0.0f; s1.f[i] = 0.0f; }
#pragma unroll
        for (int df = 0; df < 2; ++df) {
            ABFrag kb0, kb1;
            load_bt_frag(kb0, Kst + df * 32, KLD_, lane);
            load_bt_frag(kb1, Kst + 16 * KLD_ + df * 32, KLD_, lane);
            s0.v = wmma_bf16(qf[df], kb0, s0.v);
            s1.v = wmma_bf16(qf[df], kb1, s1.v);
        }

        // ---- online softmax ----
        float bm[8];
#pragma unroll
        for (int i = 0; i < 8; ++i) bm[i] = fmaxf(s0.f[i], s1.f[i]);
#pragma unroll
        for (int off = 1; off < 16; off <<= 1)
#pragma unroll
            for (int i = 0; i < 8; ++i)
                bm[i] = fmaxf(bm[i], __shfl_xor(bm[i], off, 16));

        float alpha[8], rs[8];
#pragma unroll
        for (int i = 0; i < 8; ++i) {
            float mn = fmaxf(m[i], bm[i]);
            alpha[i] = __expf(m[i] - mn);
            m[i] = mn;
            s0.f[i] = __expf(s0.f[i] - mn);
            s1.f[i] = __expf(s1.f[i] - mn);
            rs[i] = s0.f[i] + s1.f[i];
        }
#pragma unroll
        for (int off = 1; off < 16; off <<= 1)
#pragma unroll
            for (int i = 0; i < 8; ++i)
                rs[i] += __shfl_xor(rs[i], off, 16);
#pragma unroll
        for (int i = 0; i < 8; ++i) l[i] = l[i] * alpha[i] + rs[i];
#pragma unroll
        for (int f = 0; f < 4; ++f)
#pragma unroll
            for (int i = 0; i < 8; ++i) acc[f].f[i] *= alpha[i];

        // ---- transpose P (D-layout -> A-layout) via LDS, as bf16 ----
#pragma unroll
        for (int i = 0; i < 8; ++i) {
            Plds[(rbase + i) * PLD_ + col]      = f2bf(s0.f[i]);
            Plds[(rbase + i) * PLD_ + col + 16] = f2bf(s1.f[i]);
        }
        ABFrag pf;
        load_a_frag(pf, Plds, PLD_, lane);   // same-wave DS ordering

        // ---- out += P @ V ----
#pragma unroll
        for (int nb = 0; nb < 4; ++nb) {
            ABFrag vb;
            load_bt_frag(vb, Vst + (size_t)(nb * 16) * VLD_, VLD_, lane);
            acc[nb].v = wmma_bf16(pf, vb, acc[nb].v);
        }
    }

    // ---- epilogue: normalize, write concat buffer ----
#pragma unroll
    for (int nb = 0; nb < 4; ++nb)
#pragma unroll
        for (int i = 0; i < 8; ++i) {
            float o = acc[nb].f[i] / l[i];
            int srow = sblk * 16 + rbase + i;
            int c    = h * E_ + nb * 16 + col;
            att[((size_t)b * S_ + srow) * (H_ * E_) + c] = f2bf(o);
        }
}

// ---------------------------------------------------------------------------
// Output projection: out[M=8192][N=1024] = att @ Wo + bo  (fp32 out),
// register-double-buffered K-loop like qkv_proj.
// ---------------------------------------------------------------------------
__global__ __launch_bounds__(32)
void out_proj(const unsigned short* __restrict__ att,
              const unsigned short* __restrict__ WoT,   // [N][K]
              const float* __restrict__ bo,
              float* __restrict__ out) {
    const int lane = threadIdx.x & 31;
    const int mblk = blockIdx.x;
    const int nblk = blockIdx.y;
    const int K = H_ * E_;
    const int N = D_;

    const unsigned short* abase = att + (size_t)mblk * 16 * K;
    const unsigned short* wbase = WoT + (size_t)(nblk * 64) * K;

    CDFrag acc[4];
#pragma unroll
    for (int f = 0; f < 4; ++f)
#pragma unroll
        for (int i = 0; i < 8; ++i) acc[f].f[i] = 0.0f;

    ABFrag a0, a1, b0[4], b1[4];
    load_a_frag(a0, abase, K, lane);
#pragma unroll
    for (int nb = 0; nb < 4; ++nb)
        load_bt_frag(b0[nb], wbase + (size_t)(nb * 16) * K, K, lane);

    for (int k = 0; k < K; k += 64) {
        load_a_frag(a1, abase + k + 32, K, lane);
#pragma unroll
        for (int nb = 0; nb < 4; ++nb)
            load_bt_frag(b1[nb], wbase + (size_t)(nb * 16) * K + k + 32, K, lane);
#pragma unroll
        for (int nb = 0; nb < 4; ++nb) acc[nb].v = wmma_bf16(a0, b0[nb], acc[nb].v);
        if (k + 64 < K) {
            load_a_frag(a0, abase + k + 64, K, lane);
#pragma unroll
            for (int nb = 0; nb < 4; ++nb)
                load_bt_frag(b0[nb], wbase + (size_t)(nb * 16) * K + k + 64, K, lane);
        }
#pragma unroll
        for (int nb = 0; nb < 4; ++nb) acc[nb].v = wmma_bf16(a1, b1[nb], acc[nb].v);
    }

    const int col   = lane & 15;
    const int rbase = (lane >> 4) ? 8 : 0;
#pragma unroll
    for (int nb = 0; nb < 4; ++nb)
#pragma unroll
        for (int i = 0; i < 8; ++i) {
            int row = mblk * 16 + rbase + i;
            int n   = nblk * 64 + nb * 16 + col;
            out[(size_t)row * N + n] = acc[nb].f[i] + bo[n];
        }
}

// ---------------------------------------------------------------------------
// Launch
// ---------------------------------------------------------------------------
extern "C" void kernel_launch(void* const* d_in, const int* in_sizes, int n_in,
                              void* d_out, int out_size, void* d_ws, size_t ws_size,
                              hipStream_t stream) {
    const float* x  = (const float*)d_in[0];
    const float* Wq = (const float*)d_in[1];
    const float* Wk = (const float*)d_in[2];
    const float* Wv = (const float*)d_in[3];
    const float* Wo = (const float*)d_in[4];
    const float* bo = (const float*)d_in[5];
    float* out = (float*)d_out;

    char* ws = (char*)d_ws;
    size_t off = 0;
    auto carve = [&](size_t bytes) { size_t o = off; off += (bytes + 255) & ~(size_t)255; return o; };
    unsigned short* xb  = (unsigned short*)(ws + carve((size_t)B_ * S_ * D_ * 2));
    unsigned short* Wt  = (unsigned short*)(ws + carve((size_t)3 * H_ * E_ * D_ * 2));
    unsigned short* WoT = (unsigned short*)(ws + carve((size_t)D_ * H_ * E_ * 2));
    unsigned short* Qb  = (unsigned short*)(ws + carve((size_t)B_ * H_ * S_ * E_ * 2));
    unsigned short* Kb  = (unsigned short*)(ws + carve((size_t)B_ * H_ * S_ * E_ * 2));
    unsigned short* Vt  = (unsigned short*)(ws + carve((size_t)B_ * H_ * E_ * S_ * 2));
    unsigned short* att = (unsigned short*)(ws + carve((size_t)B_ * S_ * H_ * E_ * 2));

    {
        int n = B_ * S_ * D_;
        cvt_f32_to_bf16<<<(n + 255) / 256, 256, 0, stream>>>(x, xb, n);
    }
    {
        int n = H_ * D_ * E_;
        cvt_w_transpose<<<(n + 255) / 256, 256, 0, stream>>>(Wq, Wt + 0 * (size_t)n);
        cvt_w_transpose<<<(n + 255) / 256, 256, 0, stream>>>(Wk, Wt + 1 * (size_t)n);
        cvt_w_transpose<<<(n + 255) / 256, 256, 0, stream>>>(Wv, Wt + 2 * (size_t)n);
        int nw = (H_ * E_) * D_;
        cvt_wo_transpose<<<(nw + 255) / 256, 256, 0, stream>>>(Wo, WoT);
    }
    qkv_proj<<<dim3(S_ / 16, H_, 3 * B_), 32, 0, stream>>>(xb, Wt, Qb, Kb, Vt);
    flash_attn<<<dim3(S_ / 16, H_, B_), 32, 0, stream>>>(Qb, Kb, Vt, att);
    out_proj<<<dim3(B_ * S_ / 16, D_ / 64), 32, 0, stream>>>(att, WoT, bo, out);
}
